// Attention_23682449670490
// MI455X (gfx1250) — matrix operand order repaired and checked by reading, not compiled, not run on hardware
//
#include <hip/hip_runtime.h>
#include <hip/hip_bf16.h>
#include <cmath>

#define DIM_   1024
#define HEADS_ 16
#define DHEAD_ 64
#define INNER_ 1024

typedef __attribute__((ext_vector_type(16))) _Float16 v16h;
typedef __attribute__((ext_vector_type(8)))  _Float16 v8h;
typedef __attribute__((ext_vector_type(4)))  _Float16 v4h;
typedef __attribute__((ext_vector_type(8)))  float    v8f;
typedef __attribute__((ext_vector_type(4)))  unsigned int v4u;
typedef __attribute__((ext_vector_type(8)))  int v8i;
typedef __attribute__((ext_vector_type(4)))  int v4i;

#ifndef __has_builtin
#define __has_builtin(x) 0
#endif
#if defined(__HIP_DEVICE_COMPILE__) && __has_builtin(__builtin_amdgcn_tensor_load_to_lds)
#define HAVE_TDM 1
#else
#define HAVE_TDM 0
#endif

__device__ __forceinline__ v16h pack16(v8h lo, v8h hi) {
  v16h r;
#pragma unroll
  for (int i = 0; i < 8; ++i) { r[i] = lo[i]; r[i + 8] = hi[i]; }
  return r;
}
__device__ __forceinline__ v16h ld16(const _Float16* p0, const _Float16* p1) {
  return pack16(*(const v8h*)p0, *(const v8h*)p1);
}
__device__ __forceinline__ v8f wmma_f16(v16h a, v16h b, v8f c) {
  return __builtin_amdgcn_wmma_f32_16x16x32_f16(false, a, false, b, (short)0, c,
                                                false, false);
}

// Async 16B copy global -> LDS (no VGPR data path, tracked by ASYNCcnt).
// ldsOff = low 32 bits of flat shared pointer = workgroup-relative LDS addr.
__device__ __forceinline__ void async_copy_b128(unsigned int ldsOff, const void* g) {
#if defined(__HIP_DEVICE_COMPILE__)
  asm volatile("global_load_async_to_lds_b128 %0, %1, off"
               :: "v"(ldsOff), "v"(g) : "memory");
#endif
}
__device__ __forceinline__ void wait_async0() {
#if defined(__HIP_DEVICE_COMPILE__)
  asm volatile("s_wait_asynccnt 0x0" ::: "memory");
#endif
}

// TDM: DMA a 2D f16 tile (tile_dim0=64 elems/row, tile_dim1=32 rows, row
// stride 2048 elems) into LDS, padding 4 DWORDs every 32 DWORDs so LDS row
// stride becomes 72 halfs (matches sK layout).
__device__ __forceinline__ void tdm_load_k_tile(unsigned int ldsOff, const void* gp) {
#if HAVE_TDM
  const unsigned long long ga = (unsigned long long)(uintptr_t)gp;
  v4u g0;
  g0[0] = 1u;                                            // count=1, user mode
  g0[1] = ldsOff;                                        // lds_addr
  g0[2] = (unsigned int)(ga & 0xFFFFFFFFu);              // global_addr lo
  g0[3] = (unsigned int)((ga >> 32) & 0x1FFFFFFu) | (2u << 30);  // hi | type=2
  v8i g1;
  g1[0] = (int)((1u << 16)      /* data_size=2B */
              | (1u << 20)      /* pad_enable   */
              | (4u << 22)      /* pad_interval: 32 DWORDs */
              | (3u << 25));    /* pad_amount:   4 DWORDs  */
  g1[1] = (int)(2048u << 16);   // tensor_dim0 = 2048 (elems)
  g1[2] = (int)(4096u << 16);   // tensor_dim1 = 4096 rows
  g1[3] = (int)(64u << 16);     // tile_dim0 = 64
  g1[4] = 32;                   // tile_dim1 = 32, tile_dim2 = 0
  g1[5] = 2048;                 // tensor_dim0_stride = 2048 elems
  g1[6] = 0;
  g1[7] = 0;
  const v4i z4 = {0, 0, 0, 0};
  const v8i z8 = {0, 0, 0, 0, 0, 0, 0, 0};
  __builtin_amdgcn_tensor_load_to_lds(g0, g1, z4, z4, z8, 0);
  __builtin_amdgcn_s_wait_tensorcnt(0);
#endif
}

// ---------------- LayerNorm: fp32 row -> f16 normalized row ----------------
__global__ __launch_bounds__(256) void ln_f16_kernel(
    const float* __restrict__ x, const float* __restrict__ gamma,
    const float* __restrict__ beta, _Float16* __restrict__ xn) {
  const int row = blockIdx.x;
  const int t   = threadIdx.x;
  const float4 xv = *(const float4*)(x + (size_t)row * DIM_ + t * 4);
  float s  = xv.x + xv.y + xv.z + xv.w;
  float ss = xv.x * xv.x + xv.y * xv.y + xv.z * xv.z + xv.w * xv.w;
#pragma unroll
  for (int off = 16; off >= 1; off >>= 1) {
    s  += __shfl_xor(s, off, 32);
    ss += __shfl_xor(ss, off, 32);
  }
  __shared__ float rs[8], rss[8];
  if ((t & 31) == 0) { rs[t >> 5] = s; rss[t >> 5] = ss; }
  __syncthreads();
  float tot = 0.f, tot2 = 0.f;
#pragma unroll
  for (int i = 0; i < 8; ++i) { tot += rs[i]; tot2 += rss[i]; }
  const float mu   = tot * (1.0f / DIM_);
  const float var  = tot2 * (1.0f / DIM_) - mu * mu;
  const float rstd = rsqrtf(var + 1e-5f);
  const float4 gv = *(const float4*)(gamma + t * 4);
  const float4 bv = *(const float4*)(beta + t * 4);
  v4h o;
  o[0] = (_Float16)((xv.x - mu) * rstd * gv.x + bv.x);
  o[1] = (_Float16)((xv.y - mu) * rstd * gv.y + bv.y);
  o[2] = (_Float16)((xv.z - mu) * rstd * gv.z + bv.z);
  o[3] = (_Float16)((xv.w - mu) * rstd * gv.w + bv.w);
  *(v4h*)(xn + (size_t)row * DIM_ + t * 4) = o;
}

// ------- fp32 [K][N] -> f16 transposed [N][K] (coalesced via LDS tile) -----
__global__ __launch_bounds__(256) void cvtT_f16_kernel(
    const float* __restrict__ w, _Float16* __restrict__ o, int K, int N,
    float scale) {
  __shared__ float tile[32][33];
  const int k0 = blockIdx.x * 32;
  const int n0 = blockIdx.y * 32;
  const int t  = threadIdx.x;
  const int r  = t >> 5, c = t & 31;
#pragma unroll
  for (int p = 0; p < 4; ++p)
    tile[r + p * 8][c] = w[(size_t)(k0 + r + p * 8) * N + n0 + c];
  __syncthreads();
#pragma unroll
  for (int p = 0; p < 4; ++p)
    o[(size_t)(n0 + r + p * 8) * K + k0 + c] =
        (_Float16)(tile[c][r + p * 8] * scale);
}

// ------- V slice of kv -> per-head transposed vT[(b*H+h)*64+d][n] ----------
__global__ __launch_bounds__(256) void transpose_v_kernel(
    const _Float16* __restrict__ kv, _Float16* __restrict__ vT, int n) {
  __shared__ _Float16 tile[64][72];
  const int j0   = blockIdx.x * 64;
  const int h    = blockIdx.y;
  const size_t bb = blockIdx.z;
  const int t  = threadIdx.x;
  const int jr = t >> 3;
  const int dc = (t & 7) * 8;
#pragma unroll
  for (int p = 0; p < 2; ++p) {
    const int j = jr + p * 32;
    const v8h v = *(const v8h*)(kv + (bb * n + j0 + j) * (size_t)(2 * INNER_) +
                                INNER_ + h * DHEAD_ + dc);
#pragma unroll
    for (int i = 0; i < 8; ++i) tile[dc + i][j] = v[i];
  }
  __syncthreads();
#pragma unroll
  for (int p = 0; p < 2; ++p) {
    const int d = jr + p * 32;
    *(v8h*)(vT + ((bb * HEADS_ + h) * (size_t)DHEAD_ + d) * n + j0 + dc) =
        *(const v8h*)&tile[d][dc];
  }
}

// ---------------- Tiled WMMA GEMM: C[M,N] = A[M,K] @ BT[N][K]^T -----------
// Block tile 128x64, 8 waves each own a 16x64 strip. BT tile async-copied
// into LDS (padded stride 40 halfs) with zero VGPR involvement.
__global__ __launch_bounds__(256) void gemm_wmma_kernel(
    const _Float16* __restrict__ A, const _Float16* __restrict__ BT,
    void* __restrict__ C, int M, int N, int K, int outF32) {
  __shared__ _Float16 sBT[64 * 40];  // [n][k], padded stride 40
  const int t    = threadIdx.x;
  const int wave = t >> 5;
  const int lane = t & 31;
  const int lrow = lane & 15;
  const int hi   = (lane >> 4) & 1;
  const int m0   = blockIdx.x * 128 + wave * 16;
  const int n0   = blockIdx.y * 64;
  const int nrow = t >> 2;        // staging: 0..63 (n)
  const int kc   = (t & 3) * 8;   // staging: 0,8,16,24 (k)
  const unsigned int dstOff =
      (unsigned int)(uintptr_t)&sBT[nrow * 40 + kc];

  v8f acc[4] = {};
  for (int k0 = 0; k0 < K; k0 += 32) {
    __syncthreads();
    async_copy_b128(dstOff, BT + (size_t)(n0 + nrow) * K + k0 + kc);
    wait_async0();
    __syncthreads();
    const _Float16* ap = A + (size_t)(m0 + lrow) * K + k0 + hi * 16;
    const v16h a = ld16(ap, ap + 8);
#pragma unroll
    for (int nt = 0; nt < 4; ++nt) {
      const _Float16* bp = &sBT[(nt * 16 + lrow) * 40 + hi * 16];
      const v16h b = pack16(*(const v8h*)bp, *(const v8h*)(bp + 8));
      acc[nt] = wmma_f16(a, b, acc[nt]);
    }
  }
#pragma unroll
  for (int nt = 0; nt < 4; ++nt) {
#pragma unroll
    for (int r = 0; r < 8; ++r) {
      const int row = m0 + r + hi * 8;
      const int col = n0 + nt * 16 + lrow;
      if (outF32)
        ((float*)C)[(size_t)row * N + col] = acc[nt][r];
      else
        ((_Float16*)C)[(size_t)row * N + col] = (_Float16)acc[nt][r];
    }
  }
}

// ---------------- Flash attention with WMMA + TDM/async staging -----------
// grid: (n/128, HEADS, b); block 256 = 8 waves, each wave owns 16 query rows.
__global__ __launch_bounds__(256) void attn_wmma_kernel(
    const _Float16* __restrict__ q,    // [b*n, INNER], pre-scaled
    const _Float16* __restrict__ kv,   // [b*n, 2*INNER] (k | v)
    const _Float16* __restrict__ vT,   // [(b*H+h)*64+d][n]
    const float* __restrict__ bias,    // [H, n, n]
    _Float16* __restrict__ ao,         // [b*n, INNER]
    int n) {
  __shared__ _Float16 sK[32 * 72];      // K tile row-major [j][d], pad 72
  __shared__ _Float16 sVT[64 * 40];     // V tile transposed [d][j], pad 40
  __shared__ _Float16 sP[8 * 16 * 40];  // per-wave P tile [i][j], pad 40

  const int t    = threadIdx.x;
  const int wave = t >> 5;
  const int lane = t & 31;
  const int lrow = lane & 15;
  const int hi   = (lane >> 4) & 1;
  const int h    = blockIdx.y;
  const int iBase = blockIdx.x * 128;
  const int iW    = iBase + wave * 16;
  const size_t rowBase = (size_t)blockIdx.z * n;
  const size_t bh = (size_t)blockIdx.z * HEADS_ + h;
  const int koff = h * DHEAD_;

  // Q tile (16x64) kept in registers as two K=32 A operands
  const _Float16* qp = q + (rowBase + iW + lrow) * INNER_ + koff + hi * 16;
  const v16h aq0 = ld16(qp, qp + 8);
  const v16h aq1 = ld16(qp + 32, qp + 40);

  v8f o0 = {}, o1 = {}, o2 = {}, o3 = {};
  float m[8], l[8];
#pragma unroll
  for (int r = 0; r < 8; ++r) { m[r] = -1e30f; l[r] = 0.f; }

  const float* bptr = bias + (size_t)h * n * n;
  const int jEnd = (iBase + 128 < n) ? (iBase + 128) : n;  // causal bound
  _Float16* myP = &sP[wave * 16 * 40];

  // per-thread staging coords: Vt tile d-row 0..63, j-chunk 0/8/16/24
  const int vdr = t >> 2;
  const int vjc = (t & 3) * 8;
  const unsigned int vDst = (unsigned int)(uintptr_t)&sVT[vdr * 40 + vjc];
  const unsigned int kBase = (unsigned int)(uintptr_t)&sK[0];
#if !HAVE_TDM
  const int kjr = t >> 3;        // 0..31 (j)
  const int kdc = (t & 7) * 8;   // 0..56 (d)
  const unsigned int kDst = (unsigned int)(uintptr_t)&sK[kjr * 72 + kdc];
#endif

  for (int j0 = 0; j0 < jEnd; j0 += 32) {
    __syncthreads();
    // Vt tile: pure async copy, padded dest stride folded into vDst
    async_copy_b128(vDst, vT + (bh * DHEAD_ + vdr) * n + j0 + vjc);
#if HAVE_TDM
    if (wave == 0)  // one TDM descriptor moves the whole padded K tile
      tdm_load_k_tile(kBase, kv + (rowBase + j0) * (size_t)(2 * INNER_) + koff);
#else
    async_copy_b128(kDst,
                    kv + (rowBase + j0 + kjr) * (size_t)(2 * INNER_) + koff + kdc);
#endif
    wait_async0();
    __syncthreads();
    if (j0 > iW + 15) continue;  // tile fully above diagonal for this wave
    (void)kBase;

    // S = Q @ K^T : two 16x16 tiles (keys j0..15, j0+16..31), K-dim 64
    v8f s0 = {}, s1 = {};
    {
      const _Float16* bp0 = &sK[lrow * 72 + hi * 16];
      s0 = wmma_f16(aq0, pack16(*(const v8h*)bp0, *(const v8h*)(bp0 + 8)), s0);
      s0 = wmma_f16(aq1, pack16(*(const v8h*)(bp0 + 32), *(const v8h*)(bp0 + 40)), s0);
      const _Float16* bp1 = &sK[(16 + lrow) * 72 + hi * 16];
      s1 = wmma_f16(aq0, pack16(*(const v8h*)bp1, *(const v8h*)(bp1 + 8)), s1);
      s1 = wmma_f16(aq1, pack16(*(const v8h*)(bp1 + 32), *(const v8h*)(bp1 + 40)), s1);
    }

    if (j0 + 32 < jEnd)
      __builtin_prefetch(&bptr[(size_t)(iW + hi * 8) * n + j0 + 32 + lrow], 0, 1);

    // bias + causal mask + online softmax (each row lives across 16 lanes)
    float p0[8], p1[8], alpha[8];
#pragma unroll
    for (int r = 0; r < 8; ++r) {
      const int i = iW + r + hi * 8;
      float a0 = s0[r] + bptr[(size_t)i * n + j0 + lrow];
      float a1 = s1[r] + bptr[(size_t)i * n + j0 + 16 + lrow];
      if (j0 + lrow > i)      a0 = -1e30f;
      if (j0 + 16 + lrow > i) a1 = -1e30f;
      float mx = fmaxf(a0, a1);
#pragma unroll
      for (int off = 8; off >= 1; off >>= 1) mx = fmaxf(mx, __shfl_xor(mx, off, 16));
      const float mNew = fmaxf(m[r], mx);
      alpha[r] = __expf(m[r] - mNew);
      p0[r] = __expf(a0 - mNew);
      p1[r] = __expf(a1 - mNew);
      float sum = p0[r] + p1[r];
#pragma unroll
      for (int off = 8; off >= 1; off >>= 1) sum += __shfl_xor(sum, off, 16);
      l[r] = l[r] * alpha[r] + sum;
      m[r] = mNew;
    }

    // rescale O, spill P to LDS (C-layout -> memory) for A-operand reshape
#pragma unroll
    for (int r = 0; r < 8; ++r) {
      o0[r] *= alpha[r]; o1[r] *= alpha[r]; o2[r] *= alpha[r]; o3[r] *= alpha[r];
      const int prow = r + hi * 8;
      myP[prow * 40 + lrow]      = (_Float16)p0[r];
      myP[prow * 40 + 16 + lrow] = (_Float16)p1[r];
    }
    const _Float16* pp = &myP[lrow * 40 + hi * 16];
    const v16h aP = ld16(pp, pp + 8);  // A layout 16x32

    // O += P @ V : 4 WMMAs over head-dim tiles, B from transposed V in LDS
    {
      const _Float16* vp0 = &sVT[(0  + lrow) * 40 + hi * 16];
      o0 = wmma_f16(aP, pack16(*(const v8h*)vp0, *(const v8h*)(vp0 + 8)), o0);
      const _Float16* vp1 = &sVT[(16 + lrow) * 40 + hi * 16];
      o1 = wmma_f16(aP, pack16(*(const v8h*)vp1, *(const v8h*)(vp1 + 8)), o1);
      const _Float16* vp2 = &sVT[(32 + lrow) * 40 + hi * 16];
      o2 = wmma_f16(aP, pack16(*(const v8h*)vp2, *(const v8h*)(vp2 + 8)), o2);
      const _Float16* vp3 = &sVT[(48 + lrow) * 40 + hi * 16];
      o3 = wmma_f16(aP, pack16(*(const v8h*)vp3, *(const v8h*)(vp3 + 8)), o3);
    }
  }

  // normalize by row-sum and store f16
#pragma unroll
  for (int r = 0; r < 8; ++r) {
    const float inv = 1.0f / l[r];
    const int i = iW + r + hi * 8;
    _Float16* op = ao + (rowBase + i) * INNER_ + koff;
    op[lrow]      = (_Float16)(o0[r] * inv);
    op[16 + lrow] = (_Float16)(o1[r] * inv);
    op[32 + lrow] = (_Float16)(o2[r] * inv);
    op[48 + lrow] = (_Float16)(o3[r] * inv);
  }
}

// ---------------------------------------------------------------------------
extern "C" void kernel_launch(void* const* d_in, const int* in_sizes, int n_in,
                              void* d_out, int out_size, void* d_ws,
                              size_t ws_size, hipStream_t stream) {
  const float* x     = (const float*)d_in[0];
  const float* bias  = (const float*)d_in[1];
  const float* gamma = (const float*)d_in[2];
  const float* beta  = (const float*)d_in[3];
  const float* wq    = (const float*)d_in[4];
  const float* wkv   = (const float*)d_in[5];
  const float* wo    = (const float*)d_in[6];

  const long nn = (long)in_sizes[1] / HEADS_;        // n*n
  const int  n  = (int)(std::sqrt((double)nn) + 0.5);
  const int  rows = (int)((long)in_sizes[0] / ((long)n * DIM_));  // b*n
  const int  bsz  = rows / n;

  _Float16* p = (_Float16*)d_ws;
  _Float16* xn    = p; p += (size_t)rows * DIM_;
  _Float16* wqT   = p; p += (size_t)INNER_ * DIM_;       // [N][K]
  _Float16* wkvT  = p; p += (size_t)2 * INNER_ * DIM_;   // [N][K]
  _Float16* woT   = p; p += (size_t)DIM_ * INNER_;       // [N][K]
  _Float16* qb    = p; p += (size_t)rows * INNER_;
  _Float16* kvb   = p; p += (size_t)rows * 2 * INNER_;
  _Float16* vTb   = p; p += (size_t)rows * INNER_;
  _Float16* aob   = p;

  ln_f16_kernel<<<rows, 256, 0, stream>>>(x, gamma, beta, xn);

  const float qscale = 0.125f;  // DIM_HEAD^-0.5, folded into wq
  cvtT_f16_kernel<<<dim3(DIM_ / 32, INNER_ / 32), 256, 0, stream>>>(
      wq, wqT, DIM_, INNER_, qscale);
  cvtT_f16_kernel<<<dim3(DIM_ / 32, 2 * INNER_ / 32), 256, 0, stream>>>(
      wkv, wkvT, DIM_, 2 * INNER_, 1.0f);
  cvtT_f16_kernel<<<dim3(INNER_ / 32, DIM_ / 32), 256, 0, stream>>>(
      wo, woT, INNER_, DIM_, 1.0f);

  dim3 gq(rows / 128, INNER_ / 64);
  gemm_wmma_kernel<<<gq, 256, 0, stream>>>(xn, wqT, qb, rows, INNER_, DIM_, 0);
  dim3 gkv(rows / 128, 2 * INNER_ / 64);
  gemm_wmma_kernel<<<gkv, 256, 0, stream>>>(xn, wkvT, kvb, rows, 2 * INNER_, DIM_, 0);

  transpose_v_kernel<<<dim3(n / 64, HEADS_, bsz), 256, 0, stream>>>(kvb, vTb, n);

  dim3 ga(n / 128, HEADS_, bsz);
  attn_wmma_kernel<<<ga, 256, 0, stream>>>(qb, kvb, vTb, bias, aob, n);

  dim3 go(rows / 128, DIM_ / 64);
  gemm_wmma_kernel<<<go, 256, 0, stream>>>(aob, woT, d_out, rows, DIM_, INNER_, 1);
}
